// DST_Decoder_62251255989118
// MI455X (gfx1250) — compile-verified, matrix operand order checked by reading, not run on hardware
//
#include <hip/hip_runtime.h>
#include <hip/hip_bf16.h>

// ---------------- CDNA5 WMMA helpers (wave32, bf16 16x16x32, f32 accum) ----
typedef __attribute__((ext_vector_type(16))) __bf16 v16bf;
typedef __attribute__((ext_vector_type(8)))  __bf16 v8bf;
typedef __attribute__((ext_vector_type(8)))  float  v8f;

__device__ __forceinline__ v8f wmma_bf16(v16bf a, v16bf b, v8f c) {
    // (neg_a, A, neg_b, B, c_mod, C, reuse_a, reuse_b)
    return __builtin_amdgcn_wmma_f32_16x16x32_bf16(false, a, false, b, (short)0, c,
                                                   false, false);
}

// Load a 16x32 bf16 tile fragment from LDS (row-major, `stride` bf16 per row).
// Per ISA: lane<16 -> M=lane, K = {0..7, 16..23}; lane>=16 -> M=lane-16,
// K = {8..15, 24..31}. Same pattern serves B when stored as B^T (N-major).
__device__ __forceinline__ v16bf ldfrag(const __bf16* base, int stride, int lane) {
    int m  = lane & 15;
    int kb = (lane >> 4) << 3;            // 0 or 8
    const __bf16* p = base + m * stride + kb;
    v8bf lo = *(const v8bf*)(p);
    v8bf hi = *(const v8bf*)(p + 16);
    v16bf r;
#pragma unroll
    for (int i = 0; i < 8; ++i) { r[i] = lo[i]; r[i + 8] = hi[i]; }
    return r;
}

// Problem constants
#define BATCH 32
#define SEQ   1024
#define DIN   256
#define CH    32        // augmented channels
#define HID   64
#define DOUT  128
#define T     128       // chunk length
#define KC    8         // chunks per batch
#define NCHUNK (BATCH * KC)

// ============================ Kernel 1 =====================================
// Per chunk: h = relu(x@w1+b1); p = h@w2+b2; dx (basepoint diff);
// within-chunk inclusive cumsum S1loc; chunk dx totals.
__global__ __launch_bounds__(256) void k1_augment_dx(
        const float* __restrict__ x,  const float* __restrict__ w1,
        const float* __restrict__ b1, const float* __restrict__ w2,
        const float* __restrict__ b2,
        float* __restrict__ dxg, float* __restrict__ s1g, float* __restrict__ csDx)
{
    __shared__ __align__(16) __bf16 Xs[T * 32];      // x K-slice (bf16)
    __shared__ __align__(16) __bf16 W1s[64 * 32];    // w1^T K-slice
    __shared__ __align__(16) __bf16 Hs[T * 64];      // hidden (bf16)
    __shared__ __align__(16) __bf16 W2s[32 * 64];    // w2^T
    __shared__ __align__(16) float  Ps[(T + 1) * 32];// p rows, row 0 = prev
    __shared__ float b1s[64], b2s[32], hprev[64];

    const int cid  = blockIdx.x;
    const int b    = cid >> 3, kck = cid & 7;
    const int row0 = b * SEQ + kck * T;
    const int tid  = threadIdx.x;
    const int lane = tid & 31, w = tid >> 5;

    if (tid < 64) b1s[tid] = b1[tid];
    if (tid < 32) b2s[tid] = b2[tid];
    for (int i = tid; i < 32 * 64; i += 256) {       // W2s[c][d] = w2[d][c]
        int c = i >> 6, d = i & 63;
        W2s[i] = (__bf16)w2[d * 32 + c];
    }
    // p_prev row (row0-1) recomputed scalar; zero for chunk 0 (basepoint)
    if (tid < 64) {
        float acc = b1[tid];
        if (kck > 0) {
            const float* xr = x + (size_t)(row0 - 1) * DIN;
            for (int j = 0; j < DIN; ++j) acc += xr[j] * w1[j * 64 + tid];
        }
        hprev[tid] = (kck > 0) ? fmaxf(acc, 0.f) : 0.f;
    }
    __syncthreads();
    if (tid < 32) {
        float acc = 0.f;
        if (kck > 0) {
            acc = b2s[tid];
            for (int d = 0; d < 64; ++d) acc += hprev[d] * w2[d * 32 + tid];
        }
        Ps[tid] = acc;
    }

    // GEMM1: (128 x 256) x (256 x 64), K streamed in slices of 32
    v8f acc1[4] = {};
    for (int k0 = 0; k0 < DIN / 32; ++k0) {
        __syncthreads();
        for (int i = tid; i < T * 32; i += 256) {
            int s = i >> 5, j = i & 31;
            Xs[i] = (__bf16)x[(size_t)(row0 + s) * DIN + k0 * 32 + j];
        }
        for (int i = tid; i < 64 * 32; i += 256) {
            int n = i >> 5, kk = i & 31;
            W1s[i] = (__bf16)w1[(k0 * 32 + kk) * 64 + n];
        }
        __syncthreads();
        v16bf a = ldfrag(&Xs[(w * 16) * 32], 32, lane);
#pragma unroll
        for (int ct = 0; ct < 4; ++ct) {
            v16bf bm = ldfrag(&W1s[(ct * 16) * 32], 32, lane);
            acc1[ct] = wmma_bf16(a, bm, acc1[ct]);
        }
    }
    __syncthreads();
    const int mhi = (lane >> 4) << 3;
#pragma unroll
    for (int ct = 0; ct < 4; ++ct)
#pragma unroll
        for (int r = 0; r < 8; ++r) {
            int m = w * 16 + mhi + r, n = ct * 16 + (lane & 15);
            Hs[m * 64 + n] = (__bf16)fmaxf(acc1[ct][r] + b1s[n], 0.f);
        }
    __syncthreads();

    // GEMM2: (128 x 64) x (64 x 32)
    v8f acc2[2] = {};
#pragma unroll
    for (int ks = 0; ks < 2; ++ks) {
        v16bf a = ldfrag(&Hs[(w * 16) * 64 + ks * 32], 64, lane);
#pragma unroll
        for (int ct = 0; ct < 2; ++ct) {
            v16bf bm = ldfrag(&W2s[(ct * 16) * 64 + ks * 32], 64, lane);
            acc2[ct] = wmma_bf16(a, bm, acc2[ct]);
        }
    }
#pragma unroll
    for (int ct = 0; ct < 2; ++ct)
#pragma unroll
        for (int r = 0; r < 8; ++r) {
            int m = w * 16 + mhi + r, n = ct * 16 + (lane & 15);
            Ps[(1 + m) * 32 + n] = acc2[ct][r] + b2s[n];
        }
    __syncthreads();

    // dx + within-chunk inclusive scan (32 channels, serial over 128 rows)
    if (tid < 32) {
        float acc = 0.f;
        for (int s = 0; s < T; ++s) {
            float d = Ps[(s + 1) * 32 + tid] - Ps[s * 32 + tid];
            acc += d;
            dxg[(size_t)(row0 + s) * 32 + tid] = d;
            s1g[(size_t)(row0 + s) * 32 + tid] = acc;
        }
        csDx[cid * 32 + tid] = acc;
    }
}

// ============================ Kernel 2 =====================================
// Per chunk: build virtual sig rows r_s = [S1glob_s | vec(u_s (x) dx_s)]
// one 32-wide K-slice at a time, WMMA against W1 (1056x64) with split
// accumulators (S1 part uncumsummed -> Pa; outer part -> g, scanned -> Pg).
__global__ __launch_bounds__(256) void k2_sig_head1(
        const float* __restrict__ dxg, const float* __restrict__ s1g,
        const float* __restrict__ csDx, const float* __restrict__ W1g,
        float* __restrict__ Pa, float* __restrict__ Pg, float* __restrict__ csG)
{
    __shared__ __align__(16) char sm[46080];
    float*  Dxs = (float*)sm;              // [128][32]
    float*  U   = (float*)(sm + 16384);    // [128][32]  u_s = S1prev + 0.5 dx
    __bf16* Rk  = (__bf16*)(sm + 32768);   // [128][32]  current K-slice (bf16)
    __bf16* Ws  = (__bf16*)(sm + 40960);   // [64][32]   W1^T K-slice
    float*  offA = (float*)(sm + 45056);   // [32]
    float*  Gs  = (float*)sm;              // reuse: [128][64] g rows

    const int cid  = blockIdx.x;
    const int b    = cid >> 3, kck = cid & 7;
    const int row0 = b * SEQ + kck * T;
    const int tid  = threadIdx.x;
    const int lane = tid & 31, w = tid >> 5;

    if (tid < 32) {                        // cross-chunk S1 offset
        float o = 0.f;
        for (int j = 0; j < kck; ++j) o += csDx[(b * KC + j) * 32 + tid];
        offA[tid] = o;
    }
    __syncthreads();
    for (int i = tid; i < T * 32; i += 256) {
        int s = i >> 5, c = i & 31;
        float d  = dxg[(size_t)(row0 + s) * 32 + c];
        float sl = s1g[(size_t)(row0 + s) * 32 + c];
        Dxs[i] = d;
        U[i]   = offA[c] + sl - 0.5f * d;  // u_s ; S1glob_s = U + 0.5 dx
    }

    v8f accA[4] = {}, accG[4] = {};
    for (int kt = 0; kt < 33; ++kt) {
        __syncthreads();
        for (int i = tid; i < T * 32; i += 256) {
            int s = i >> 5, j = i & 31;
            float v = (kt == 0) ? (U[i] + 0.5f * Dxs[i])
                                : (U[s * 32 + (kt - 1)] * Dxs[s * 32 + j]);
            Rk[i] = (__bf16)v;
        }
        for (int i = tid; i < 64 * 32; i += 256) {
            int n = i >> 5, kk = i & 31;
            Ws[i] = (__bf16)W1g[(size_t)(kt * 32 + kk) * 64 + n];
        }
        if (kt < 32) __builtin_prefetch(W1g + (size_t)((kt + 1) * 32) * 64, 0, 0);
        __syncthreads();
        v16bf a = ldfrag(&Rk[(w * 16) * 32], 32, lane);
#pragma unroll
        for (int ct = 0; ct < 4; ++ct) {
            v16bf bm = ldfrag(&Ws[(ct * 16) * 32], 32, lane);
            if (kt == 0) accA[ct] = wmma_bf16(a, bm, accA[ct]);
            else         accG[ct] = wmma_bf16(a, bm, accG[ct]);
        }
    }
    __syncthreads();

    const int mhi = (lane >> 4) << 3;
#pragma unroll
    for (int ct = 0; ct < 4; ++ct)                   // a rows -> global
#pragma unroll
        for (int r = 0; r < 8; ++r) {
            int m = w * 16 + mhi + r, n = ct * 16 + (lane & 15);
            Pa[(size_t)(row0 + m) * 64 + n] = accA[ct][r];
            Gs[m * 64 + n] = accG[ct][r];            // g rows -> LDS (reused)
        }
    __syncthreads();
    if (tid < 64) {                                  // 64-ch scan of g
        float acc = 0.f;
        for (int s = 0; s < T; ++s) {
            acc += Gs[s * 64 + tid];
            Pg[(size_t)(row0 + s) * 64 + tid] = acc;
        }
        csG[cid * 64 + tid] = acc;
    }
}

// ============================ Kernel 3 =====================================
// Per chunk: pre1 = Pa + Pg + offG + bb1; relu; (128x64)x(64x128) GEMM -> out
__global__ __launch_bounds__(256) void k3_head2(
        const float* __restrict__ Pa, const float* __restrict__ Pg,
        const float* __restrict__ csG, const float* __restrict__ bb1,
        const float* __restrict__ W2g, const float* __restrict__ bb2,
        float* __restrict__ out)
{
    __shared__ __align__(16) __bf16 Zs[T * 64];      // relu'd hidden (bf16)
    __shared__ __align__(16) __bf16 W2s[128 * 64];   // W2^T [n][k]
    __shared__ float offG[64], b1s[64], b2s[128];

    const int cid  = blockIdx.x;
    const int b    = cid >> 3, kck = cid & 7;
    const int row0 = b * SEQ + kck * T;
    const int tid  = threadIdx.x;
    const int lane = tid & 31, w = tid >> 5;

    if (tid < 64) {
        float o = 0.f;
        for (int j = 0; j < kck; ++j) o += csG[(b * KC + j) * 64 + tid];
        offG[tid] = o;
        b1s[tid]  = bb1[tid];
    }
    if (tid < 128) b2s[tid] = bb2[tid];
    for (int i = tid; i < 128 * 64; i += 256) {      // W2s[n][k] = W2[k][n]
        int n = i >> 6, kk = i & 63;
        W2s[i] = (__bf16)W2g[kk * 128 + n];
    }
    __syncthreads();
    for (int i = tid; i < T * 64; i += 256) {
        int s = i >> 6, c = i & 63;
        float v = Pa[(size_t)(row0 + s) * 64 + c] + Pg[(size_t)(row0 + s) * 64 + c]
                + offG[c] + b1s[c];
        Zs[i] = (__bf16)fmaxf(v, 0.f);
    }
    __syncthreads();

    const int mhi = (lane >> 4) << 3;
    for (int ct = 0; ct < 8; ++ct) {
        v8f acc = {};
#pragma unroll
        for (int ks = 0; ks < 2; ++ks) {
            v16bf a  = ldfrag(&Zs[(w * 16) * 64 + ks * 32], 64, lane);
            v16bf bm = ldfrag(&W2s[(ct * 16) * 64 + ks * 32], 64, lane);
            acc = wmma_bf16(a, bm, acc);
        }
#pragma unroll
        for (int r = 0; r < 8; ++r) {
            int m = w * 16 + mhi + r, n = ct * 16 + (lane & 15);
            out[(size_t)(row0 + m) * DOUT + n] = acc[r] + b2s[n];
        }
    }
}

// ============================ Launch =======================================
extern "C" void kernel_launch(void* const* d_in, const int* in_sizes, int n_in,
                              void* d_out, int out_size, void* d_ws, size_t ws_size,
                              hipStream_t stream) {
    const float* x   = (const float*)d_in[0];
    const float* w1  = (const float*)d_in[1];
    const float* b1  = (const float*)d_in[2];
    const float* w2  = (const float*)d_in[3];
    const float* b2  = (const float*)d_in[4];
    const float* W1  = (const float*)d_in[5];
    const float* bb1 = (const float*)d_in[6];
    const float* W2  = (const float*)d_in[7];
    const float* bb2 = (const float*)d_in[8];
    float* out = (float*)d_out;

    const size_t N = (size_t)BATCH * SEQ;            // 32768 rows
    float* dxg  = (float*)d_ws;                      // N*32
    float* s1g  = dxg + N * 32;                      // N*32
    float* csDx = s1g + N * 32;                      // NCHUNK*32
    float* Pa   = csDx + (size_t)NCHUNK * 32;        // N*64
    float* Pg   = Pa + N * 64;                       // N*64
    float* csG  = Pg + N * 64;                       // NCHUNK*64
    (void)ws_size; (void)in_sizes; (void)n_in; (void)out_size;

    k1_augment_dx<<<NCHUNK, 256, 0, stream>>>(x, w1, b1, w2, b2, dxg, s1g, csDx);
    k2_sig_head1 <<<NCHUNK, 256, 0, stream>>>(dxg, s1g, csDx, W1, Pa, Pg, csG);
    k3_head2     <<<NCHUNK, 256, 0, stream>>>(Pa, Pg, csG, bb1, W2, bb2, out);
}
